// MoAKDALayer_77000173682734
// MI455X (gfx1250) — matrix operand order repaired and checked by reading, not compiled, not run on hardware
//
#include <hip/hip_runtime.h>
#include <hip/hip_bf16.h>
#include <math.h>

typedef __attribute__((ext_vector_type(16))) _Float16 v16h;
typedef __attribute__((ext_vector_type(8)))  _Float16 v8h;
typedef __attribute__((ext_vector_type(8)))  float    v8f;

#define B_    4
#define T_    1024
#define D_    512
#define DK_   16
#define DV_   16
#define KP_   32
#define H_    4
#define EQ_   8
#define EK_   6
#define N_    4
#define HQ_   32
#define HK_   24
#define R_    4
#define DA_   25
#define ND_   2048
#define CD_   64
#define BT_   4096      /* B*T tokens */
#define ROWS_ 16384     /* B*T*H rows */
#define GN_   224       /* gate gemm N = (HQ+HK)*N */
#define RN_   64        /* router gemm N (56 padded to 64) */
#define PG_   64        /* DPG=39 padded to 64 */
#define DPG_  39
#define TWO_PI_ 6.283185307179586f

__device__ __forceinline__ float wred32(float x) {
#pragma unroll
  for (int o = 16; o > 0; o >>= 1) x += __shfl_xor(x, o, 32);
  return x;
}
__device__ __forceinline__ float sigm(float x) { return 1.f / (1.f + __expf(-x)); }
__device__ __forceinline__ float siluf(float x) { return x * sigm(x); }

// ---------------------------------------------------------------------------
// Generic WMMA GEMM: C[M,N](f32) = A[M,K](f16 row-major) x Bt[N,K](f16, B^T)
// One 16x16 tile per wave; K multiple of 32, N multiple of 16, M multiple of 16.
// A frag (ISA 7.12.2, 16-bit A 16x32): lane<16 holds K[0..7]+K[16..23] of row M=lane;
// lane>=16 holds K[8..15]+K[24..31]. B frag: same K split as A, for col N.
// ---------------------------------------------------------------------------
__global__ void gemm16_kernel(const _Float16* __restrict__ A,
                              const _Float16* __restrict__ Bt,
                              float* __restrict__ C, int M, int N, int K) {
  int lane = threadIdx.x & 31;
  int wid = blockIdx.x * (blockDim.x >> 5) + (threadIdx.x >> 5);
  int ntiles = N >> 4;
  int total = (M >> 4) * ntiles;
  if (wid >= total) return;
  int mt = wid / ntiles, nt = wid - mt * ntiles;
  int hw = lane >> 4;                    // half-wave
  const _Float16* arow = A + (size_t)(mt * 16 + (lane & 15)) * K + (hw << 3);
  const _Float16* brow = Bt + (size_t)(nt * 16 + (lane & 15)) * K + (hw << 3);
  v8f acc = {};
  for (int k0 = 0; k0 < K; k0 += 32) {
    __builtin_prefetch(arow + k0 + 128, 0, 1);
    __builtin_prefetch(brow + k0 + 128, 0, 1);
    v8h alo = *(const v8h*)(arow + k0);
    v8h ahi = *(const v8h*)(arow + k0 + 16);
    v16h av = __builtin_shufflevector(alo, ahi, 0,1,2,3,4,5,6,7,8,9,10,11,12,13,14,15);
    v8h blo = *(const v8h*)(brow + k0);
    v8h bhi = *(const v8h*)(brow + k0 + 16);
    v16h bv = __builtin_shufflevector(blo, bhi, 0,1,2,3,4,5,6,7,8,9,10,11,12,13,14,15);
    acc = __builtin_amdgcn_wmma_f32_16x16x32_f16(false, av, false, bv,
                                                 (short)0, acc, false, false);
  }
  int n = nt * 16 + (lane & 15);
  int mb = mt * 16 + (hw << 3);
#pragma unroll
  for (int r = 0; r < 8; ++r) C[(size_t)(mb + r) * N + n] = acc[r];
}

// ---------------------------------------------------------------------------
// Weight packing: out[n*Kout+k] = in[k*Nin+n] (f32->f16, zero-padded)
// ---------------------------------------------------------------------------
__global__ void pack_transpose_kernel(const float* __restrict__ in,
                                      _Float16* __restrict__ out,
                                      int Kout, int Nout, int Kin, int Nin) {
  int i = blockIdx.x * blockDim.x + threadIdx.x;
  if (i >= Kout * Nout) return;
  int n = i / Kout, k = i - n * Kout;
  float v = (n < Nin && k < Kin) ? in[(size_t)k * Nin + n] : 0.f;
  out[i] = (_Float16)v;
}

// Fold RMS weight into gate phases: row r<128 -> q expert e=r/4, n=r%4; else kv.
__global__ void pack_gate_kernel(const float* __restrict__ mq_phi, const float* __restrict__ mq_norm,
                                 const float* __restrict__ mkv_phi, const float* __restrict__ mkv_norm,
                                 _Float16* __restrict__ G) {
  size_t i = (size_t)blockIdx.x * blockDim.x + threadIdx.x;
  if (i >= (size_t)GN_ * ND_) return;
  int r = (int)(i / ND_), c = (int)(i - (size_t)r * ND_);
  float v;
  if (r < HQ_ * N_) {
    int e = r >> 2, n = r & 3;
    v = mq_phi[((size_t)e * N_ + n) * ND_ + c] * mq_norm[(size_t)e * ND_ + c];
  } else {
    int rr = r - HQ_ * N_;
    int e = rr >> 2, n = rr & 3;
    v = mkv_phi[((size_t)e * N_ + n) * ND_ + c] * mkv_norm[(size_t)e * ND_ + c];
  }
  G[i] = (_Float16)v;
}

__global__ void pack_routers_kernel(const float* __restrict__ qr, const float* __restrict__ kr,
                                    _Float16* __restrict__ Rp) {
  int i = blockIdx.x * blockDim.x + threadIdx.x;
  if (i >= RN_ * D_) return;
  int r = i / D_, c = i - r * D_;
  float v = (r < H_ * EQ_) ? qr[(size_t)r * D_ + c]
          : (r < H_ * EQ_ + H_ * EK_) ? kr[(size_t)(r - H_ * EQ_) * D_ + c] : 0.f;
  Rp[i] = (_Float16)v;
}

// ---------------------------------------------------------------------------
// Token stats: shared RMS norm (xh f16) + route = mean over streams (f16)
// one block per token, 256 threads, 2048 elements
// ---------------------------------------------------------------------------
__global__ void stats_kernel(const float* __restrict__ stream,
                             _Float16* __restrict__ xh16, _Float16* __restrict__ route16) {
  int bt = blockIdx.x;
  int b = bt >> 10, t = bt & (T_ - 1);
  int tid = threadIdx.x;
  float sv[8]; float ss = 0.f;
#pragma unroll
  for (int n = 0; n < N_; n++)
#pragma unroll
    for (int j = 0; j < 2; j++) {
      float v = stream[(((size_t)(b * N_ + n) * T_ + t) * D_) + tid + 256 * j];
      sv[n * 2 + j] = v; ss += v * v;
    }
  __shared__ float red[256];
  red[tid] = ss; __syncthreads();
  for (int s = 128; s > 0; s >>= 1) { if (tid < s) red[tid] += red[tid + s]; __syncthreads(); }
  float inv = rsqrtf(red[0] / (float)ND_ + 1e-6f);
#pragma unroll
  for (int n = 0; n < N_; n++)
#pragma unroll
    for (int j = 0; j < 2; j++) {
      int d = tid + 256 * j;
      xh16[(size_t)bt * ND_ + n * D_ + d] = (_Float16)(sv[n * 2 + j] * inv);
    }
#pragma unroll
  for (int j = 0; j < 2; j++) {
    int d = tid + 256 * j;
    float r = 0.25f * (sv[j] + sv[2 + j] + sv[4 + j] + sv[6 + j]);
    route16[(size_t)bt * D_ + d] = (_Float16)r;
  }
}

// top-1 routing per head (q: expert0 masked), coeff = max softmax
__global__ void select_kernel(const float* __restrict__ logits,
                              int* __restrict__ qsel, float* __restrict__ qcoef,
                              int* __restrict__ ksel, float* __restrict__ kcoef) {
  int i = blockIdx.x * blockDim.x + threadIdx.x;
  if (i >= BT_ * H_) return;
  int bt = i >> 2, h = i & 3;
  const float* L = logits + (size_t)bt * RN_;
  {
    float mx = -1e30f; int mi = 0;
    for (int e = 0; e < EQ_; e++) { float v = L[h * EQ_ + e]; if (v > mx) { mx = v; mi = e; } }
    float s = 0.f; for (int e = 0; e < EQ_; e++) s += __expf(L[h * EQ_ + e] - mx);
    qsel[i] = mi; qcoef[i] = (mi > 0) ? (1.f / s) : 0.f;   // expert 0 masked out
  }
  {
    const float* Lk = L + H_ * EQ_;
    float mx = -1e30f; int mi = 0;
    for (int e = 0; e < EK_; e++) { float v = Lk[h * EK_ + e]; if (v > mx) { mx = v; mi = e; } }
    float s = 0.f; for (int e = 0; e < EK_; e++) s += __expf(Lk[h * EK_ + e] - mx);
    ksel[i] = mi; kcoef[i] = 1.f / s;
  }
}

// ---------------------------------------------------------------------------
// Selected-expert h_e = rms( sum_n sigmoid(a*Hp+b)[n] * stream[b,n,t,:], norm )
// one block per token; builds hq16/hkv16 (ROWS x 512, f16) + routed sums
// ---------------------------------------------------------------------------
__global__ void combine_kernel(const float* __restrict__ stream, const float* __restrict__ hp_raw,
                               const int* __restrict__ qsel, const int* __restrict__ ksel,
                               const float* __restrict__ mq_a, const float* __restrict__ mq_b,
                               const float* __restrict__ mkv_a, const float* __restrict__ mkv_b,
                               const float* __restrict__ norm_q, const float* __restrict__ norm_kv,
                               _Float16* __restrict__ hq16, _Float16* __restrict__ hkv16,
                               _Float16* __restrict__ hqr16, _Float16* __restrict__ hkvr16) {
  int bt = blockIdx.x;
  int b = bt >> 10, t = bt & (T_ - 1);
  int tid = threadIdx.x;
  float sv[8];
#pragma unroll
  for (int n = 0; n < N_; n++)
#pragma unroll
    for (int j = 0; j < 2; j++)
      sv[n * 2 + j] = stream[(((size_t)(b * N_ + n) * T_ + t) * D_) + tid + 256 * j];
  __shared__ float red[256];
  float rq[2] = {0.f, 0.f}, rkv[2] = {0.f, 0.f};
  for (int s = 0; s < 8; s++) {
    int h = s & 3; bool isq = (s < 4);
    int e = isq ? qsel[bt * H_ + h] : ksel[bt * H_ + h];
    int eg = isq ? (h * EQ_ + e) : (h * EK_ + e);
    int col = isq ? eg : (H_ * EQ_ + eg);
    float ap = isq ? mq_a[eg] : mkv_a[eg];
    float hp[4];
#pragma unroll
    for (int n = 0; n < N_; n++) {
      float raw = hp_raw[(size_t)bt * GN_ + col * N_ + n];
      float bb = isq ? mq_b[eg * N_ + n] : mkv_b[eg * N_ + n];
      hp[n] = sigm(ap * raw + bb);
    }
    float he[2]; float ss = 0.f;
#pragma unroll
    for (int j = 0; j < 2; j++) {
      float a = hp[0] * sv[j] + hp[1] * sv[2 + j] + hp[2] * sv[4 + j] + hp[3] * sv[6 + j];
      he[j] = a; ss += a * a;
    }
    red[tid] = ss; __syncthreads();
    for (int r = 128; r > 0; r >>= 1) { if (tid < r) red[tid] += red[tid + r]; __syncthreads(); }
    float inv = rsqrtf(red[0] / (float)D_ + 1e-6f);
    __syncthreads();
#pragma unroll
    for (int j = 0; j < 2; j++) {
      int d = tid + 256 * j;
      float w = isq ? norm_q[(size_t)eg * D_ + d] : norm_kv[(size_t)eg * D_ + d];
      float v = he[j] * inv * w;
      size_t rowoff = ((size_t)bt * H_ + h) * D_ + d;
      if (isq) { hq16[rowoff] = (_Float16)v; if (e > 0) rq[j] += v; }
      else     { hkv16[rowoff] = (_Float16)v; rkv[j] += v; }
    }
  }
#pragma unroll
  for (int j = 0; j < 2; j++) {
    int d = tid + 256 * j;
    hqr16[(size_t)bt * D_ + d]  = (_Float16)(rq[j]  * 0.25f);
    hkvr16[(size_t)bt * D_ + d] = (_Float16)(rkv[j] * 0.25f);
  }
}

// ---------------------------------------------------------------------------
// Q epilogue: lora delta + l2norm + PoPE; one wave per (b,t,h) row
// qbuf layout (B,H,T,KP)
// ---------------------------------------------------------------------------
__global__ void q_epilogue_kernel(const float* __restrict__ qlin, const _Float16* __restrict__ hq16,
                                  const int* __restrict__ qsel, const float* __restrict__ qcoef,
                                  const float* __restrict__ loraA, const float* __restrict__ loraB,
                                  float* __restrict__ qbuf) {
  int wid = blockIdx.x * (blockDim.x >> 5) + (threadIdx.x >> 5);
  int lane = threadIdx.x & 31;
  int bt = wid >> 2, h = wid & 3;
  int b = bt >> 10, t = bt & (T_ - 1);
  int eq = h * EQ_ + qsel[bt * H_ + h];
  float coeff = qcoef[bt * H_ + h];
  const _Float16* hp = hq16 + (size_t)wid * D_;
  float acc[R_] = {0.f, 0.f, 0.f, 0.f};
#pragma unroll 4
  for (int i = 0; i < 16; i++) {
    int d = lane + 32 * i;
    float hv = (float)hp[d];
    const float* Ap = loraA + ((size_t)eq * D_ + d) * R_;
#pragma unroll
    for (int r = 0; r < R_; r++) acc[r] += hv * Ap[r];
  }
  float u[R_];
#pragma unroll
  for (int r = 0; r < R_; r++) u[r] = siluf(wred32(acc[r]));
  int c = lane & 15;
  float dq = 0.f;
#pragma unroll
  for (int r = 0; r < R_; r++) dq += u[r] * loraB[((size_t)eq * R_ + r) * DK_ + c];
  float qv = qlin[(size_t)wid * DK_ + c] + dq;
  float sq = (lane < 16) ? qv * qv : 0.f;
  float nrm = sqrtf(wred32(sq));
  float qn = qv / fmaxf(nrm, 1e-12f);
  float mu = log1pf(__expf(qn));               // softplus
  float fr = __powf(10000.f, (float)c / (float)DK_);
  float phi = (float)t * fr;
  if (lane < 16) {
    size_t rb = (((size_t)b * H_ + h) * T_ + t) * KP_;
    qbuf[rb + c]       = coeff * mu * __cosf(phi);
    qbuf[rb + DK_ + c] = coeff * mu * __sinf(phi);
  }
}

// ---------------------------------------------------------------------------
// KV epilogue: loras + alpha/beta MLPs + PoPE(k); one wave per row; h_e in LDS
// ---------------------------------------------------------------------------
__global__ void kv_epilogue_kernel(const float* __restrict__ kvlin, const _Float16* __restrict__ hkv16,
                                   const int* __restrict__ ksel, const float* __restrict__ kcoef,
                                   const float* __restrict__ pope_delta,
                                   const float* __restrict__ lAk, const float* __restrict__ lBk,
                                   const float* __restrict__ lAv, const float* __restrict__ lBv,
                                   const float* __restrict__ aup, const float* __restrict__ adn,
                                   const float* __restrict__ bup, const float* __restrict__ bdn,
                                   float* __restrict__ kbuf, float* __restrict__ vbuf,
                                   float* __restrict__ abuf, float* __restrict__ bbuf) {
  int w = threadIdx.x >> 5;
  int wid = blockIdx.x * (blockDim.x >> 5) + w;
  int lane = threadIdx.x & 31;
  int bt = wid >> 2, h = wid & 3;
  int b = bt >> 10, t = bt & (T_ - 1);
  int ek = h * EK_ + ksel[bt * H_ + h];
  float coeff = kcoef[bt * H_ + h];
  __shared__ float hl[4][D_];
  const _Float16* hp = hkv16 + (size_t)wid * D_;
#pragma unroll 4
  for (int i = 0; i < 16; i++) hl[w][lane + 32 * i] = (float)hp[lane + 32 * i];
  __syncthreads();
  // lora k/v
  float ak[R_] = {0,0,0,0}, av[R_] = {0,0,0,0};
#pragma unroll 4
  for (int i = 0; i < 16; i++) {
    int d = lane + 32 * i;
    float hv = hl[w][d];
    const float* Akp = lAk + ((size_t)ek * D_ + d) * R_;
    const float* Avp = lAv + ((size_t)ek * D_ + d) * R_;
#pragma unroll
    for (int r = 0; r < R_; r++) { ak[r] += hv * Akp[r]; av[r] += hv * Avp[r]; }
  }
  float uk[R_], uv[R_];
#pragma unroll
  for (int r = 0; r < R_; r++) { uk[r] = siluf(wred32(ak[r])); uv[r] = siluf(wred32(av[r])); }
  int c = lane & 15;
  float dk = 0.f, dv = 0.f;
#pragma unroll
  for (int r = 0; r < R_; r++) {
    dk += uk[r] * lBk[((size_t)ek * R_ + r) * DK_ + c];
    dv += uv[r] * lBv[((size_t)ek * R_ + r) * DV_ + c];
  }
  // alpha/beta up-projections: lane l<25 computes column l
  float aau = 0.f, abu = 0.f;
  for (int d = 0; d < D_; d++) {
    float hv = hl[w][d];
    if (lane < DA_) {
      aau += hv * aup[((size_t)ek * D_ + d) * DA_ + lane];
      abu += hv * bup[((size_t)ek * D_ + d) * DA_ + lane];
    }
  }
  float ua = (lane < DA_) ? siluf(aau) : 0.f;
  float ub = (lane < DA_) ? siluf(abu) : 0.f;
  float accA = 0.f;
  for (int l = 0; l < DA_; l++) {
    float us = __shfl(ua, l, 32);
    accA += us * adn[((size_t)ek * DA_ + l) * KP_ + lane];
  }
  float a_e = sigm(accA);                                    // decay (col = lane, 32 cols)
  float bpart = (lane < DA_) ? ub * bdn[(size_t)ek * DA_ + lane] : 0.f;
  float b_e = sigm(wred32(bpart));
  // k PoPE + v
  float kv_ = kvlin[(size_t)wid * KP_ + c] + dk;             // cols 0..15 = W_k path
  float vv  = kvlin[(size_t)wid * KP_ + DK_ + c] + dv;       // cols 16..31 = W_v path
  float sqk = (lane < 16) ? kv_ * kv_ : 0.f;
  float nk = sqrtf(wred32(sqk));
  float kn = kv_ / fmaxf(nk, 1e-12f);
  float muk = log1pf(__expf(kn));
  float fr = __powf(10000.f, (float)c / (float)DK_);
  float phik = (float)t * fr - TWO_PI_ * sigm(pope_delta[c]);
  size_t rb = ((size_t)b * H_ + h) * T_ + t;
  if (lane < 16) {
    kbuf[rb * KP_ + c]       = muk * __cosf(phik);
    kbuf[rb * KP_ + DK_ + c] = muk * __sinf(phik);
    vbuf[rb * DV_ + c] = vv * coeff;
  }
  abuf[rb * KP_ + lane] = a_e;
  if (lane == 0) bbuf[rb] = b_e;
}

// ---------------------------------------------------------------------------
// KDA delta-rule scan with S-mHC mixing. block = (b,h); wave = e (DV), lane = d (KP)
// State S[i][d][e], i=0..3, in registers (4 floats per thread). xs==0 since S0==0.
// ---------------------------------------------------------------------------
__global__ void kda_scan_kernel(const float* __restrict__ qbuf, const float* __restrict__ kbuf,
                                const float* __restrict__ vbuf, const float* __restrict__ abuf,
                                const float* __restrict__ bbuf,
                                const float* __restrict__ b_pre, const float* __restrict__ b_post,
                                const float* __restrict__ b_res,
                                float* __restrict__ obuf, float* __restrict__ Sout) {
  int blk = blockIdx.x;                 // b*H + h
  int b = blk >> 2, h = blk & 3;
  int w = threadIdx.x >> 5;             // e
  int lane = threadIdx.x & 31;          // d
  float Hpre[N_], Hpost[N_], Hres[N_ * N_];
#pragma unroll
  for (int i = 0; i < N_; i++) {
    Hpre[i]  = sigm(b_pre[h * N_ + i]);
    Hpost[i] = 2.f * sigm(b_post[h * N_ + i]);
  }
#pragma unroll
  for (int i = 0; i < N_ * N_; i++) Hres[i] = __expf(b_res[h * N_ * N_ + i]);
  for (int it = 0; it < 6; it++) {
#pragma unroll
    for (int i = 0; i < N_; i++) {
      float s = Hres[i*4] + Hres[i*4+1] + Hres[i*4+2] + Hres[i*4+3];
#pragma unroll
      for (int j = 0; j < N_; j++) Hres[i*4+j] /= s;
    }
#pragma unroll
    for (int j = 0; j < N_; j++) {
      float s = Hres[j] + Hres[4+j] + Hres[8+j] + Hres[12+j];
#pragma unroll
      for (int i = 0; i < N_; i++) Hres[i*4+j] /= s;
    }
  }
  float S0 = 0.f, S1 = 0.f, S2 = 0.f, S3 = 0.f;
  size_t base = (size_t)blk * T_;
  for (int t = 0; t < T_; t++) {
    size_t r = base + t;
    float q = qbuf[r * KP_ + lane];
    float k = kbuf[r * KP_ + lane];
    float a = abuf[r * KP_ + lane];
    float v = vbuf[r * DV_ + w];
    float bb = bbuf[r];
    float s  = Hpre[0]*S0 + Hpre[1]*S1 + Hpre[2]*S2 + Hpre[3]*S3;
    float aS = a * s;
    float kt = wred32(k * aS);          // k . (a*s) over d
    float sn = aS + bb * k * (v - kt);
    float o  = wred32(q * sn);          // q . s_new over d
    if (lane == 0) obuf[((size_t)b * T_ + t) * CD_ + h * DV_ + w] = o;
    float dlt = sn - s;
    float n0 = Hres[0]*S0 + Hres[1]*S1 + Hres[2]*S2 + Hres[3]*S3   + Hpost[0]*dlt;
    float n1 = Hres[4]*S0 + Hres[5]*S1 + Hres[6]*S2 + Hres[7]*S3   + Hpost[1]*dlt;
    float n2 = Hres[8]*S0 + Hres[9]*S1 + Hres[10]*S2 + Hres[11]*S3 + Hpost[2]*dlt;
    float n3 = Hres[12]*S0 + Hres[13]*S1 + Hres[14]*S2 + Hres[15]*S3 + Hpost[3]*dlt;
    S0 = n0; S1 = n1; S2 = n2; S3 = n3;
  }
  // S_new (H,B,N,KP,DV)
  size_t sb = (((size_t)h * B_ + b) * N_) * KP_ * DV_;
  Sout[sb + ((size_t)0 * KP_ + lane) * DV_ + w] = S0;
  Sout[sb + ((size_t)1 * KP_ + lane) * DV_ + w] = S1;
  Sout[sb + ((size_t)2 * KP_ + lane) * DV_ + w] = S2;
  Sout[sb + ((size_t)3 * KP_ + lane) * DV_ + w] = S3;
}

// o * sigmoid(pre-gate) -> f16 for W_o GEMM
__global__ void gate_o_kernel(const float* __restrict__ o, const float* __restrict__ pre,
                              _Float16* __restrict__ og16) {
  int i = blockIdx.x * blockDim.x + threadIdx.x;
  if (i >= BT_ * CD_) return;
  og16[i] = (_Float16)(o[i] * sigm(pre[i]));
}

__global__ void silu_g1_kernel(const float* __restrict__ g1, _Float16* __restrict__ g1h) {
  int i = blockIdx.x * blockDim.x + threadIdx.x;
  if (i >= BT_ * PG_) return;
  g1h[i] = (_Float16)siluf(g1[i]);
}

// y = ylin * sigmoid(g2); broadcast over N streams
__global__ void finalize_kernel(const float* __restrict__ ylin, const float* __restrict__ g2,
                                float* __restrict__ out) {
  int i = blockIdx.x * blockDim.x + threadIdx.x;
  if (i >= BT_ * D_) return;
  int bt = i / D_, d = i - bt * D_;
  int b = bt >> 10, t = bt & (T_ - 1);
  float y = ylin[i] * sigm(g2[i]);
#pragma unroll
  for (int n = 0; n < N_; n++)
    out[(((size_t)(b * N_ + n) * T_ + t) * D_) + d] = y;
}

// ---------------------------------------------------------------------------
extern "C" void kernel_launch(void* const* d_in, const int* in_sizes, int n_in,
                              void* d_out, int out_size, void* d_ws, size_t ws_size,
                              hipStream_t stream) {
  (void)in_sizes; (void)n_in; (void)out_size; (void)ws_size;
  const float* s_in   = (const float*)d_in[0];
  const float* W_q    = (const float*)d_in[1];
  const float* W_k    = (const float*)d_in[2];
  const float* W_v    = (const float*)d_in[3];
  const float* pdelta = (const float*)d_in[4];
  const float* q_rt   = (const float*)d_in[5];
  const float* kv_rt  = (const float*)d_in[6];
  const float* lAq    = (const float*)d_in[7];
  const float* lBq    = (const float*)d_in[8];
  const float* mq_n   = (const float*)d_in[9];
  const float* mq_phi = (const float*)d_in[10];
  const float* mq_b   = (const float*)d_in[11];
  const float* mq_a   = (const float*)d_in[12];
  const float* n_q    = (const float*)d_in[13];
  const float* lAk    = (const float*)d_in[14];
  const float* lBk    = (const float*)d_in[15];
  const float* lAv    = (const float*)d_in[16];
  const float* lBv    = (const float*)d_in[17];
  const float* aup    = (const float*)d_in[18];
  const float* adn    = (const float*)d_in[19];
  const float* bup    = (const float*)d_in[20];
  const float* bdn    = (const float*)d_in[21];
  const float* mkv_n  = (const float*)d_in[22];
  const float* mkv_phi= (const float*)d_in[23];
  const float* mkv_b  = (const float*)d_in[24];
  const float* mkv_a  = (const float*)d_in[25];
  const float* n_kv   = (const float*)d_in[26];
  const float* sb_pre = (const float*)d_in[31];
  const float* sb_post= (const float*)d_in[32];
  const float* sb_res = (const float*)d_in[33];
  const float* W_pre  = (const float*)d_in[35];
  const float* W_o    = (const float*)d_in[36];
  const float* W_pg1  = (const float*)d_in[37];
  const float* W_pg2  = (const float*)d_in[38];

  float* out = (float*)d_out;
  float* Sout = out + (size_t)B_ * N_ * T_ * D_;

  char* ws = (char*)d_ws;
  size_t off = 0;
  auto alloc = [&](size_t bytes) -> char* {
    off = (off + 255) & ~(size_t)255;
    char* p = ws + off; off += bytes; return p;
  };
  _Float16* xh16   = (_Float16*)alloc((size_t)BT_ * ND_ * 2);
  _Float16* route16= (_Float16*)alloc((size_t)BT_ * D_ * 2);
  float*    hp_raw = (float*)   alloc((size_t)BT_ * GN_ * 4);
  float*    logits = (float*)   alloc((size_t)BT_ * RN_ * 4);
  int*      qsel   = (int*)     alloc((size_t)BT_ * H_ * 4);
  int*      ksel   = (int*)     alloc((size_t)BT_ * H_ * 4);
  float*    qcoef  = (float*)   alloc((size_t)BT_ * H_ * 4);
  float*    kcoef  = (float*)   alloc((size_t)BT_ * H_ * 4);
  _Float16* hq16   = (_Float16*)alloc((size_t)ROWS_ * D_ * 2);
  _Float16* hkv16  = (_Float16*)alloc((size_t)ROWS_ * D_ * 2);
  _Float16* hqr16  = (_Float16*)alloc((size_t)BT_ * D_ * 2);
  _Float16* hkvr16 = (_Float16*)alloc((size_t)BT_ * D_ * 2);
  float*    qlin   = (float*)   alloc((size_t)ROWS_ * DK_ * 4);
  float*    kvlin  = (float*)   alloc((size_t)ROWS_ * KP_ * 4);
  float*    qbuf   = (float*)   alloc((size_t)ROWS_ * KP_ * 4);
  float*    kbuf   = (float*)   alloc((size_t)ROWS_ * KP_ * 4);
  float*    abuf   = (float*)   alloc((size_t)ROWS_ * KP_ * 4);
  float*    vbuf   = (float*)   alloc((size_t)ROWS_ * DV_ * 4);
  float*    bbuf   = (float*)   alloc((size_t)ROWS_ * 4);
  float*    obuf   = (float*)   alloc((size_t)BT_ * CD_ * 4);
  float*    prebuf = (float*)   alloc((size_t)BT_ * CD_ * 4);
  _Float16* og16   = (_Float16*)alloc((size_t)BT_ * CD_ * 2);
  float*    ylin   = (float*)   alloc((size_t)BT_ * D_ * 4);
  float*    g1raw  = (float*)   alloc((size_t)BT_ * PG_ * 4);
  _Float16* g1h16  = (_Float16*)alloc((size_t)BT_ * PG_ * 2);
  float*    g2raw  = (float*)   alloc((size_t)BT_ * D_ * 4);
  _Float16* Gpack  = (_Float16*)alloc((size_t)GN_ * ND_ * 2);
  _Float16* Rpack  = (_Float16*)alloc((size_t)RN_ * D_ * 2);
  _Float16* Wq_t   = (_Float16*)alloc((size_t)DK_ * D_ * 2);
  _Float16* Wkv_t  = (_Float16*)alloc((size_t)KP_ * D_ * 2);
  _Float16* Wpre_t = (_Float16*)alloc((size_t)CD_ * D_ * 2);
  _Float16* Wo_t   = (_Float16*)alloc((size_t)D_ * CD_ * 2);
  _Float16* Wpg1_t = (_Float16*)alloc((size_t)PG_ * D_ * 2);
  _Float16* Wpg2_t = (_Float16*)alloc((size_t)D_ * PG_ * 2);

  auto packT = [&](const float* in, _Float16* o, int Kout, int Nout, int Kin, int Nin) {
    int tot = Kout * Nout;
    pack_transpose_kernel<<<(tot + 255) / 256, 256, 0, stream>>>(in, o, Kout, Nout, Kin, Nin);
  };
  // weight prep
  packT(W_q,  Wq_t,          D_, DK_, D_, DK_);
  packT(W_k,  Wkv_t,         D_, DK_, D_, DK_);
  packT(W_v,  Wkv_t + (size_t)DK_ * D_, D_, DK_, D_, DV_);
  packT(W_pre, Wpre_t,       D_, CD_, D_, CD_);
  packT(W_o,  Wo_t,          CD_, D_, CD_, D_);
  packT(W_pg1, Wpg1_t,       D_, PG_, D_, DPG_);
  packT(W_pg2, Wpg2_t,       PG_, D_, DPG_, D_);
  pack_gate_kernel<<<((GN_ * ND_) + 255) / 256, 256, 0, stream>>>(mq_phi, mq_n, mkv_phi, mkv_n, Gpack);
  pack_routers_kernel<<<((RN_ * D_) + 255) / 256, 256, 0, stream>>>(q_rt, kv_rt, Rpack);

  // token stats
  stats_kernel<<<BT_, 256, 0, stream>>>(s_in, xh16, route16);

  auto gemm = [&](const _Float16* A, const _Float16* Bt, float* C, int M, int N, int K) {
    int waves = (M / 16) * (N / 16);
    gemm16_kernel<<<(waves + 3) / 4, 128, 0, stream>>>(A, Bt, C, M, N, K);
  };
  // routing
  gemm(route16, Rpack, logits, BT_, RN_, D_);
  select_kernel<<<(BT_ * H_ + 255) / 256, 256, 0, stream>>>(logits, qsel, qcoef, ksel, kcoef);
  // expert gates (all experts; gathered per selection in combine)
  gemm(xh16, Gpack, hp_raw, BT_, GN_, ND_);
  combine_kernel<<<BT_, 256, 0, stream>>>(s_in, hp_raw, qsel, ksel, mq_a, mq_b, mkv_a, mkv_b,
                                          n_q, n_kv, hq16, hkv16, hqr16, hkvr16);
  // projections
  gemm(hq16,  Wq_t,  qlin,  ROWS_, DK_, D_);
  gemm(hkv16, Wkv_t, kvlin, ROWS_, KP_, D_);
  q_epilogue_kernel<<<ROWS_ / 4, 128, 0, stream>>>(qlin, hq16, qsel, qcoef, lAq, lBq, qbuf);
  kv_epilogue_kernel<<<ROWS_ / 4, 128, 0, stream>>>(kvlin, hkv16, ksel, kcoef, pdelta,
                                                    lAk, lBk, lAv, lBv, aup, adn, bup, bdn,
                                                    kbuf, vbuf, abuf, bbuf);
  // recurrence
  kda_scan_kernel<<<B_ * H_, 512, 0, stream>>>(qbuf, kbuf, vbuf, abuf, bbuf,
                                               sb_pre, sb_post, sb_res, obuf, Sout);
  // output path
  gemm(hkvr16, Wpre_t, prebuf, BT_, CD_, D_);
  gate_o_kernel<<<(BT_ * CD_ + 255) / 256, 256, 0, stream>>>(obuf, prebuf, og16);
  gemm(og16, Wo_t, ylin, BT_, D_, CD_);
  gemm(hqr16, Wpg1_t, g1raw, BT_, PG_, D_);
  silu_g1_kernel<<<(BT_ * PG_ + 255) / 256, 256, 0, stream>>>(g1raw, g1h16);
  gemm(g1h16, Wpg2_t, g2raw, BT_, D_, PG_);
  finalize_kernel<<<(BT_ * D_ + 255) / 256, 256, 0, stream>>>(ylin, g2raw, out);
}
